// PerlinAttention_78271484003148
// MI455X (gfx1250) — matrix operand order repaired and checked by reading, not compiled, not run on hardware
//
#include <hip/hip_runtime.h>
#include <hip/hip_bf16.h>
#include <math.h>

// ---------------- problem constants ----------------
#define N_   8
#define H_   12
#define T_   2048
#define HID_ 64
#define TM_  128
#define F_   266          // performer features
#define FP_  288          // padded to multiple of 32 (WMMA K step) / 16 (tiles)
#define NH_  (N_*H_)
#define NHT_ (NH_*T_)
#define E_   64           // min(T, HID)

#define KP1_  128         // conv1 K = 12*9 = 108 -> pad 128
#define KP48_ 448         // res-conv K = 48*9 = 432 -> pad 448

#define DN_    0.35355339059327373f   // 64^-0.25
#define RATIO_ 0.06131393394849658f   // 266^-0.5

typedef __attribute__((ext_vector_type(16))) __bf16 v16bf;
typedef __attribute__((ext_vector_type(8)))  float  v8f;

// ---------------- WMMA helpers (wave32, gfx1250 layouts) ----------------
__device__ __forceinline__ v8f zero8() {
  v8f z;
#pragma unroll
  for (int i = 0; i < 8; ++i) z[i] = 0.f;
  return z;
}

__device__ __forceinline__ v8f wmma_bf16(v16bf a, v16bf b, v8f c) {
  // D = A(16x32) * B(32x16) + C, f32 accumulate
  return __builtin_amdgcn_wmma_f32_16x16x32_bf16(false, a, false, b, (short)0, c,
                                                 false, false);
}

// A fragment: A row-major [16 x K], lda in elements, take K-slab [kbase, kbase+32).
__device__ __forceinline__ v16bf ldA(const __bf16* A, int lda, int lane, int kbase) {
  const int m  = lane & 15;
  const int hk = (lane & 16) ? 8 : 0;
  const __bf16* r = A + (size_t)m * lda + kbase;
  v16bf a;
#pragma unroll
  for (int e = 0; e < 8; ++e) { a[e] = r[hk + e]; a[8 + e] = r[16 + hk + e]; }
  return a;
}

// B fragment from B^T stored row-major [16 x K] (N-major).
__device__ __forceinline__ v16bf ldBT(const __bf16* BT, int ldb, int lane, int kbase) {
  const int n  = lane & 15;
  const int kb = (lane & 16) ? 8 : 0;
  const __bf16* r = BT + (size_t)n * ldb + kbase + kb;
  v16bf b;
#pragma unroll
  for (int e = 0; e < 8; ++e) { b[e] = r[e]; b[8 + e] = r[16 + e]; }
  return b;
}

// C/D 16x16 f32: lanes 0-15 -> N=lane, M=r; lanes 16-31 -> N=lane-16, M=8+r.
__device__ __forceinline__ void stD(float* D, int ldd, int lane, v8f c) {
  const int n  = lane & 15;
  const int mb = (lane & 16) ? 8 : 0;
#pragma unroll
  for (int r = 0; r < 8; ++r) D[(size_t)(mb + r) * ldd + n] = c[r];
}

// ---------------- prep kernels ----------------
__global__ void iy_kernel(const float* __restrict__ mask, float* __restrict__ iy) {
  const int n = blockIdx.x;
  if (threadIdx.x != 0) return;
  const float* m = mask + (size_t)n * T_;
  float ss = 0.f;
  for (int t = 0; t < T_; ++t) ss += (m[t] > -1.f) ? 1.f : 0.f;
  float cs = 0.f;
  for (int t = 0; t < T_; ++t) {
    cs += (m[t] > -1.f) ? 1.f : 0.f;
    float gy = (cs - 1.f) / (ss - 1.f + 1e-8f) * 2.f - 1.f;
    iy[(size_t)n * T_ + t] = ((gy + 1.f) * (float)E_ - 1.f) * 0.5f;
  }
}

__global__ void smallprep_kernel(const float* __restrict__ proj,
                                 const float* __restrict__ encw,
                                 const float* __restrict__ decw,
                                 __bf16* __restrict__ projT,
                                 __bf16* __restrict__ encT,
                                 __bf16* __restrict__ decT) {
  const int i  = blockIdx.x * blockDim.x + threadIdx.x;
  const int n1 = FP_ * HID_;
  const int n2 = n1 + 128 * 192;
  const int n3 = n2 + 128 * 128;
  if (i < n1) {
    int f = i / HID_, k = i % HID_;
    projT[i] = (__bf16)((f < F_) ? proj[f * HID_ + k] : 0.f);
  } else if (i < n2) {
    int j = i - n1; int r = j / 192, k = j % 192;
    encT[j] = (__bf16)encw[k * 128 + r];          // enc_w^T [128 x 192]
  } else if (i < n3) {
    int j = i - n2; int r = j / 128, k = j % 128;
    decT[j] = (__bf16)decw[k * 128 + r];          // dec_w^T [128 x 128]
  }
}

// conv weights -> N-major bf16 with K padding.  [co][ci][ky][kx] flattens to co*K + k.
__global__ void convw_kernel(const float* __restrict__ c1w,
                             const float* __restrict__ r1aw, const float* __restrict__ r1bw,
                             const float* __restrict__ r2aw, const float* __restrict__ r2bw,
                             __bf16* __restrict__ wc1,
                             __bf16* __restrict__ w1a, __bf16* __restrict__ w1b,
                             __bf16* __restrict__ w2a, __bf16* __restrict__ w2b) {
  const int i   = blockIdx.x * blockDim.x + threadIdx.x;
  const int nc1 = 48 * KP1_;
  const int nr  = 48 * KP48_;
  if (i < nc1) {
    const int co = i / KP1_, k = i % KP1_;
    wc1[i] = (__bf16)((k < 108) ? c1w[co * 108 + k] : 0.f);
  } else if (i < nc1 + 4 * nr) {
    const int j = i - nc1, which = j / nr, jj = j % nr;
    const int co = jj / KP48_, k = jj % KP48_;
    const float* src = (which == 0) ? r1aw : (which == 1) ? r1bw : (which == 2) ? r2aw : r2bw;
    __bf16* dst = (which == 0) ? w1a : (which == 1) ? w1b : (which == 2) ? w2a : w2b;
    dst[jj] = (__bf16)((k < 432) ? src[co * 432 + k] : 0.f);
  }
}

__global__ void rowprep_kernel(const float* __restrict__ qa, const float* __restrict__ ka,
                               const float* __restrict__ v,  const float* __restrict__ mask,
                               __bf16* __restrict__ qdn, __bf16* __restrict__ kdn,
                               __bf16* __restrict__ vbf,
                               float* __restrict__ qdiag, float* __restrict__ kdiag) {
  size_t r = (size_t)blockIdx.x * blockDim.x + threadIdx.x;
  if (r >= (size_t)NHT_) return;
  const int t = (int)(r % T_);
  const int n = (int)((r / T_) / H_);
  const bool pad = mask[(size_t)n * T_ + t] < -1.f;
  const float* qr = qa + r * HID_;
  const float* kr = ka + r * HID_;
  const float* vr = v  + r * HID_;
  float sq = 0.f, sk = 0.f;
#pragma unroll 4
  for (int j = 0; j < HID_; ++j) {
    float qv = qr[j] * DN_; sq += qv * qv; qdn[r * HID_ + j] = (__bf16)qv;
    float kv = kr[j] * DN_; sk += kv * kv; kdn[r * HID_ + j] = (__bf16)kv;
    vbf[r * HID_ + j] = (__bf16)(pad ? 0.f : vr[j]);
  }
  qdiag[r] = 0.5f * sq;
  kdiag[r] = 0.5f * sk;
}

// va_cat^T: [NH, 128, T] bf16.  rows 0..63 = grid-sampled identity, 64..127 = va.
__global__ void vat_kernel(const float* __restrict__ va, const float* __restrict__ mask,
                           const float* __restrict__ iy, __bf16* __restrict__ vaT) {
  size_t idx = (size_t)blockIdx.x * blockDim.x + threadIdx.x;
  if (idx >= (size_t)NH_ * 128 * T_) return;
  const int t  = (int)(idx % T_);
  const int e  = (int)((idx / T_) % 128);
  const int nh = (int)(idx / ((size_t)128 * T_));
  const int n  = nh / H_;
  float val;
  if (mask[(size_t)n * T_ + t] < -1.f) {
    val = 0.f;
  } else if (e < HID_) {
    const float iyv = iy[(size_t)n * T_ + t];
    const float gx  = (float)e / (float)(HID_ - 1) * 2.f - 1.f;
    const float ixv = ((gx + 1.f) * (float)E_ - 1.f) * 0.5f;
    const float y0 = floorf(iyv), x0 = floorf(ixv);
    const float wy = iyv - y0,   wx = ixv - x0;
    float o = 0.f;
#pragma unroll
    for (int dy = 0; dy < 2; ++dy)
#pragma unroll
      for (int dx = 0; dx < 2; ++dx) {
        const float yi = y0 + dy, xi = x0 + dx;
        const float wv = (dy ? wy : 1.f - wy) * (dx ? wx : 1.f - wx);
        if (yi == xi && yi >= 0.f && yi < (float)E_ && xi >= 0.f && xi < (float)E_) o += wv;
      }
    val = o;
  } else {
    val = va[((size_t)nh * T_ + t) * HID_ + (e - HID_)];
  }
  vaT[idx] = (__bf16)val;
}

// ---------------- FAVOR+ feature projection (WMMA) ----------------
__global__ void proj_kernel(const __bf16* __restrict__ qdn, const __bf16* __restrict__ kdn,
                            const __bf16* __restrict__ projT,
                            const float* __restrict__ qdiag, const float* __restrict__ kdiag,
                            __bf16* __restrict__ qp, __bf16* __restrict__ kpT) {
  __shared__ float fd[16][FP_];
  __shared__ float pmax[16][8];
  __shared__ float rmax[16];
  const int tid  = threadIdx.x, lane = tid & 31;
  const int wave = __builtin_amdgcn_readfirstlane((int)(threadIdx.x >> 5));
  const int r0  = blockIdx.x * 16;          // global row (nh*T + t0)
  const int isK = blockIdx.y;
  const __bf16* A = (isK ? kdn : qdn) + (size_t)r0 * HID_;

  v8f acc[5];
#pragma unroll
  for (int i = 0; i < 5; ++i) acc[i] = zero8();

#pragma unroll
  for (int ks = 0; ks < 2; ++ks) {
    const int kb = ks * 32;
    v16bf a = ldA(A, HID_, lane, kb);
#pragma unroll
    for (int i = 0; i < 5; ++i) {
      const int tn = wave + 4 * i;   // scalar -> s_cbranch, EXEC untouched
      if (tn < FP_ / 16) {
        v16bf b = ldBT(projT + (size_t)(tn * 16) * HID_, HID_, lane, kb);
        acc[i] = wmma_bf16(a, b, acc[i]);
      }
    }
  }
#pragma unroll
  for (int i = 0; i < 5; ++i) {
    const int tn = wave + 4 * i;
    if (tn < FP_ / 16) stD(&fd[0][tn * 16], FP_, lane, acc[i]);
  }
  __syncthreads();

  const int m = tid >> 3, s = tid & 7;
  float mx = -3.4e38f;
  for (int f = s; f < F_; f += 8) mx = fmaxf(mx, fd[m][f]);
  pmax[m][s] = mx;
  __syncthreads();
  if (s == 0) {
    float v = pmax[m][0];
#pragma unroll
    for (int j = 1; j < 8; ++j) v = fmaxf(v, pmax[m][j]);
    rmax[m] = v;
  }
  __syncthreads();

  const float diag = (isK ? kdiag : qdiag)[r0 + m];
  const float rm   = rmax[m];
  const int nh = r0 / T_;
  const int t  = (r0 % T_) + m;
  for (int f = s; f < FP_; f += 8) {
    float val = 0.f;
    if (f < F_) val = RATIO_ * (expf(fd[m][f] - diag - rm) + 1e-4f);
    if (isK) kpT[((size_t)nh * FP_ + f) * T_ + t] = (__bf16)val;
    else     qp[(size_t)(r0 + m) * FP_ + f]       = (__bf16)val;
  }
}

// ksum[nh,f] = sum_t kp
__global__ void ksum_kernel(const __bf16* __restrict__ kpT, float* __restrict__ ksum) {
  __shared__ float red[256];
  const int row = blockIdx.x;             // nh*FP + f
  const __bf16* p = kpT + (size_t)row * T_;
  float s = 0.f;
  for (int t = threadIdx.x; t < T_; t += 256) s += (float)p[t];
  red[threadIdx.x] = s;
  __syncthreads();
  for (int off = 128; off > 0; off >>= 1) {
    if (threadIdx.x < off) red[threadIdx.x] += red[threadIdx.x + off];
    __syncthreads();
  }
  if (threadIdx.x == 0) ksum[row] = red[0];
}

// ctx = kp^T @ va_cat : async-LDS double-buffered A tile, WMMA over K=2048.
#define CTX_KSTEPS (T_ / 32)
__global__ void ctx_kernel(const __bf16* __restrict__ kpT, const __bf16* __restrict__ vaT,
                           __bf16* __restrict__ ctxT) {
  __shared__ __bf16 sA[2][16 * 32];
  const int tid  = threadIdx.x;
  const int lane = tid & 31;
  const int wave = __builtin_amdgcn_readfirstlane((int)(threadIdx.x >> 5));
  const int nh = blockIdx.x, tf = blockIdx.y;
  const int te = blockIdx.z * 4 + wave;
  const __bf16* A = kpT + ((size_t)nh * FP_ + tf * 16) * T_;
  const __bf16* B = vaT + ((size_t)nh * 128 + te * 16) * T_;

  const int srow = tid >> 3;          // 0..15 (A tile row)
  const int scol = (tid & 7) * 4;     // 4 bf16 (8 bytes) per thread

  v8f acc = zero8();
  {
    unsigned dst = (unsigned)(size_t)&sA[0][srow * 32 + scol];
    const __bf16* src = A + (size_t)srow * T_ + scol;
    asm volatile("global_load_async_to_lds_b64 %0, %1, off"
                 :: "v"(dst), "v"(src) : "memory");
  }
  for (int ks = 0; ks < CTX_KSTEPS; ++ks) {
    const int cur = ks & 1;
    if (ks + 1 < CTX_KSTEPS) {
      unsigned dst = (unsigned)(size_t)&sA[cur ^ 1][srow * 32 + scol];
      const __bf16* src = A + (size_t)srow * T_ + (ks + 1) * 32 + scol;
      asm volatile("global_load_async_to_lds_b64 %0, %1, off"
                   :: "v"(dst), "v"(src) : "memory");
      asm volatile("s_wait_asynccnt 0x1" ::: "memory");
    } else {
      asm volatile("s_wait_asynccnt 0x0" ::: "memory");
    }
    __syncthreads();
    v16bf a = ldA(&sA[cur][0], 32, lane, 0);
    v16bf b = ldBT(B, T_, lane, ks * 32);
    acc = wmma_bf16(a, b, acc);
    __syncthreads();
  }
  const int n = lane & 15, mb = (lane & 16) ? 8 : 0;
  __bf16* out = ctxT + ((size_t)nh * 128 + te * 16 + n) * FP_ + tf * 16 + mb;
#pragma unroll
  for (int r = 0; r < 8; ++r) out[r] = (__bf16)acc[r];
}

// out = qp @ ctx, scale by d_inv, concat v -> pv [NHT x 192] bf16
__global__ void outpv_kernel(const __bf16* __restrict__ qp, const __bf16* __restrict__ ctxT,
                             const float* __restrict__ ksum, const __bf16* __restrict__ vbf,
                             __bf16* __restrict__ pv) {
  __shared__ float od[16][128];
  __shared__ float red[16][8];
  __shared__ float dinv[16];
  const int tid  = threadIdx.x, lane = tid & 31;
  const int wave = __builtin_amdgcn_readfirstlane((int)(threadIdx.x >> 5));
  const int r0 = blockIdx.x * 16;
  const int nh = r0 / T_;
  const __bf16* A = qp + (size_t)r0 * FP_;

  v8f a0 = zero8(), a1 = zero8();
  for (int kb = 0; kb < FP_; kb += 32) {
    v16bf a  = ldA(A, FP_, lane, kb);
    v16bf b0 = ldBT(ctxT + ((size_t)nh * 128 + (2 * wave + 0) * 16) * FP_, FP_, lane, kb);
    a0 = wmma_bf16(a, b0, a0);
    v16bf b1 = ldBT(ctxT + ((size_t)nh * 128 + (2 * wave + 1) * 16) * FP_, FP_, lane, kb);
    a1 = wmma_bf16(a, b1, a1);
  }
  stD(&od[0][(2 * wave + 0) * 16], 128, lane, a0);
  stD(&od[0][(2 * wave + 1) * 16], 128, lane, a1);
  __syncthreads();

  const int m = tid >> 3, s = tid & 7;
  float d = 0.f;
  for (int f = s; f < FP_; f += 8)
    d += (float)A[(size_t)m * FP_ + f] * ksum[(size_t)nh * FP_ + f];
  red[m][s] = d;
  __syncthreads();
  if (s == 0) {
    float v = 0.f;
#pragma unroll
    for (int j = 0; j < 8; ++j) v += red[m][j];
    dinv[m] = 1.f / v;
  }
  __syncthreads();

  for (int j = s; j < 192; j += 8) {
    float val = (j < 128) ? od[m][j] * dinv[m]
                          : (float)vbf[(size_t)(r0 + m) * HID_ + (j - 128)];
    pv[(size_t)(r0 + m) * 192 + j] = (__bf16)val;
  }
}

// t = gelu(LN(pv @ enc_w + enc_b)); score = t @ dec_w + dec_b
__global__ void encdec_kernel(const __bf16* __restrict__ pv, const __bf16* __restrict__ encT,
                              const __bf16* __restrict__ decT,
                              const float* __restrict__ enc_b, const float* __restrict__ ln_g,
                              const float* __restrict__ ln_b, const float* __restrict__ dec_b,
                              float* __restrict__ score) {
  __shared__ float  td[16][128];
  __shared__ __bf16 tb[16][128];
  __shared__ float  red[16][8], red2[16][8];
  __shared__ float  mu_s[16], rs_s[16];
  const int tid  = threadIdx.x, lane = tid & 31;
  const int wave = __builtin_amdgcn_readfirstlane((int)(threadIdx.x >> 5));
  const int r0 = blockIdx.x * 16;
  const __bf16* A = pv + (size_t)r0 * 192;

  v8f a0 = zero8(), a1 = zero8();
  for (int kb = 0; kb < 192; kb += 32) {
    v16bf a  = ldA(A, 192, lane, kb);
    v16bf b0 = ldBT(encT + (size_t)((2 * wave + 0) * 16) * 192, 192, lane, kb);
    a0 = wmma_bf16(a, b0, a0);
    v16bf b1 = ldBT(encT + (size_t)((2 * wave + 1) * 16) * 192, 192, lane, kb);
    a1 = wmma_bf16(a, b1, a1);
  }
  stD(&td[0][(2 * wave + 0) * 16], 128, lane, a0);
  stD(&td[0][(2 * wave + 1) * 16], 128, lane, a1);
  __syncthreads();

  const int m = tid >> 3, s = tid & 7;
  float sm = 0.f, sq = 0.f;
  for (int j = s; j < 128; j += 8) {
    float x = td[m][j] + enc_b[j];
    td[m][j] = x;
    sm += x; sq += x * x;
  }
  red[m][s] = sm; red2[m][s] = sq;
  __syncthreads();
  if (s == 0) {
    float a = 0.f, b = 0.f;
#pragma unroll
    for (int j = 0; j < 8; ++j) { a += red[m][j]; b += red2[m][j]; }
    const float mu = a / 128.f;
    mu_s[m] = mu;
    rs_s[m] = rsqrtf(b / 128.f - mu * mu + 1e-5f);
  }
  __syncthreads();
  const float mu = mu_s[m], rs = rs_s[m];
  for (int j = s; j < 128; j += 8) {
    float xn = (td[m][j] - mu) * rs * ln_g[j] + ln_b[j];
    float ge = 0.5f * xn * (1.f + erff(xn * 0.70710678118f));
    tb[m][j] = (__bf16)ge;
  }
  __syncthreads();

  v8f c0 = zero8(), c1 = zero8();
  for (int kb = 0; kb < 128; kb += 32) {
    v16bf a  = ldA(&tb[0][0], 128, lane, kb);   // A fragments straight from LDS
    v16bf b0 = ldBT(decT + (size_t)((2 * wave + 0) * 16) * 128, 128, lane, kb);
    c0 = wmma_bf16(a, b0, c0);
    v16bf b1 = ldBT(decT + (size_t)((2 * wave + 1) * 16) * 128, 128, lane, kb);
    c1 = wmma_bf16(a, b1, c1);
  }
  const int n = lane & 15, mb = (lane & 16) ? 8 : 0;
#pragma unroll
  for (int r = 0; r < 8; ++r) {
    const int col0 = (2 * wave + 0) * 16 + n;
    const int col1 = (2 * wave + 1) * 16 + n;
    score[(size_t)(r0 + mb + r) * 128 + col0] = c0[r] + dec_b[col0];
    score[(size_t)(r0 + mb + r) * 128 + col1] = c1[r] + dec_b[col1];
  }
}

// ---------------- conv stack as WMMA implicit GEMM ----------------
// conv1: [N,12,2048,128] f32 --(3x3 s2, relu)--> [N,48,1024,64] bf16.
// Block = one (n,y) row of 64 px; 4 waves x (16px M-tile x 3 co N-tiles); K=108 pad 128.
__global__ void conv1w_kernel(const float* __restrict__ in, const __bf16* __restrict__ wT,
                              const float* __restrict__ bias, __bf16* __restrict__ out) {
  __shared__ __bf16 sA[2][64 * 32];
  const int tid  = threadIdx.x, lane = tid & 31;
  const int wave = __builtin_amdgcn_readfirstlane((int)(threadIdx.x >> 5));
  const int y = blockIdx.x & 1023;
  const int n = blockIdx.x >> 10;
  const float* inb = in + (size_t)n * 12 * T_ * 128;

  const int p = tid & 63;   // pixel x in [0,64)
  const int h = tid >> 6;   // k half

  auto stage = [&](int buf, int kb) {
#pragma unroll
    for (int kk = 0; kk < 16; ++kk) {
      const int k = kb + h * 16 + kk;
      float val = 0.f;
      if (k < 108) {
        const int ci = k / 9, rem = k % 9, ky = rem / 3, kx = rem % 3;
        const int yy = 2 * y + ky - 1, xx = 2 * p + kx - 1;
        if (yy >= 0 && yy < T_ && xx >= 0 && xx < 128)
          val = inb[((size_t)ci * T_ + yy) * 128 + xx];
      }
      sA[buf][p * 32 + h * 16 + kk] = (__bf16)val;
    }
  };

  v8f acc[3] = {zero8(), zero8(), zero8()};
  stage(0, 0);
  for (int ks = 0; ks < KP1_ / 32; ++ks) {
    const int cur = ks & 1;
    if (ks + 1 < KP1_ / 32) stage(cur ^ 1, (ks + 1) * 32);
    __syncthreads();
    v16bf a = ldA(&sA[cur][wave * 16 * 32], 32, lane, 0);
#pragma unroll
    for (int tn = 0; tn < 3; ++tn) {
      v16bf b = ldBT(wT + (size_t)(tn * 16) * KP1_, KP1_, lane, ks * 32);
      acc[tn] = wmma_bf16(a, b, acc[tn]);
    }
    __syncthreads();
  }
  const int nl = lane & 15, mb = (lane & 16) ? 8 : 0;
#pragma unroll
  for (int tn = 0; tn < 3; ++tn) {
    const int co = tn * 16 + nl;
    const float bb = bias[co];
    __bf16* op = out + (((size_t)n * 48 + co) * 1024 + y) * 64 + wave * 16 + mb;
#pragma unroll
    for (int r = 0; r < 8; ++r) op[r] = (__bf16)fmaxf(acc[tn][r] + bb, 0.f);
  }
}

// 48->48 3x3 s1 conv (+optional residual) + relu, bf16 planes [N,48,1024,64].
// Implicit GEMM: M=64px row, N=48, K=432 pad 448, im2col double-buffered in LDS.
__global__ void conv48w_kernel(const __bf16* __restrict__ in, const __bf16* __restrict__ res,
                               const __bf16* __restrict__ wT, const float* __restrict__ bias,
                               __bf16* __restrict__ out) {
  __shared__ __bf16 sA[2][64 * 32];
  const int tid  = threadIdx.x, lane = tid & 31;
  const int wave = __builtin_amdgcn_readfirstlane((int)(threadIdx.x >> 5));
  const int y = blockIdx.x & 1023;
  const int n = blockIdx.x >> 10;
  const __bf16* inb = in + (size_t)n * 48 * 1024 * 64;

  const int p = tid & 63;
  const int h = tid >> 6;

  auto stage = [&](int buf, int kb) {
#pragma unroll
    for (int kk = 0; kk < 16; ++kk) {
      const int k = kb + h * 16 + kk;
      __bf16 val = (__bf16)0.f;
      if (k < 432) {
        const int ci = k / 9, rem = k % 9, ky = rem / 3, kx = rem % 3;
        const int yy = y + ky - 1, xx = p + kx - 1;
        if (yy >= 0 && yy < 1024 && xx >= 0 && xx < 64)
          val = inb[((size_t)ci * 1024 + yy) * 64 + xx];
      }
      sA[buf][p * 32 + h * 16 + kk] = val;
    }
  };

  v8f acc[3] = {zero8(), zero8(), zero8()};
  stage(0, 0);
  for (int ks = 0; ks < KP48_ / 32; ++ks) {
    const int cur = ks & 1;
    if (ks + 1 < KP48_ / 32) stage(cur ^ 1, (ks + 1) * 32);
    __syncthreads();
    v16bf a = ldA(&sA[cur][wave * 16 * 32], 32, lane, 0);
#pragma unroll
    for (int tn = 0; tn < 3; ++tn) {
      v16bf b = ldBT(wT + (size_t)(tn * 16) * KP48_, KP48_, lane, ks * 32);
      acc[tn] = wmma_bf16(a, b, acc[tn]);
    }
    __syncthreads();
  }
  const int nl = lane & 15, mb = (lane & 16) ? 8 : 0;
#pragma unroll
  for (int tn = 0; tn < 3; ++tn) {
    const int co = tn * 16 + nl;
    const float bb = bias[co];
    const size_t base = (((size_t)n * 48 + co) * 1024 + y) * 64 + wave * 16 + mb;
    __bf16* op = out + base;
#pragma unroll
    for (int r = 0; r < 8; ++r) {
      float vv = acc[tn][r] + bb;
      if (res) vv += (float)res[base + r];
      op[r] = (__bf16)fmaxf(vv, 0.f);
    }
  }
}

// pixel-shuffle + c2 conv + row softmax, writes final output
__global__ void final_kernel(const __bf16* __restrict__ xfin, const float* __restrict__ w,
                             const float* __restrict__ b, float* __restrict__ out) {
  __shared__ float red[128];
  const int y  = blockIdx.x % T_;
  const int co = (blockIdx.x / T_) % 12;
  const int n  = blockIdx.x / (T_ * 12);
  const int x  = threadIdx.x;
  float acc = b[co];
  for (int ci = 0; ci < 12; ++ci) {
#pragma unroll
    for (int ky = 0; ky < 3; ++ky) {
      const int yy = y + ky - 1;
      if (yy < 0 || yy >= T_) continue;
#pragma unroll
      for (int kx = 0; kx < 3; ++kx) {
        const int xx = x + kx - 1;
        if (xx < 0 || xx >= 128) continue;
        const int ch = ci * 4 + (yy & 1) * 2 + (xx & 1);   // pixel shuffle r=2
        acc += w[((co * 12 + ci) * 3 + ky) * 3 + kx] *
               (float)xfin[(((size_t)n * 48 + ch) * 1024 + (yy >> 1)) * 64 + (xx >> 1)];
      }
    }
  }
  red[x] = acc;
  __syncthreads();
  for (int off = 64; off > 0; off >>= 1) {
    if (x < off) red[x] = fmaxf(red[x], red[x + off]);
    __syncthreads();
  }
  const float mx = red[0];
  __syncthreads();
  const float e = expf(acc - mx);
  red[x] = e;
  __syncthreads();
  for (int off = 64; off > 0; off >>= 1) {
    if (x < off) red[x] += red[x + off];
    __syncthreads();
  }
  out[(size_t)blockIdx.x * 128 + x] = e / red[0];
}

// ---------------- host launcher ----------------
extern "C" void kernel_launch(void* const* d_in, const int* in_sizes, int n_in,
                              void* d_out, int out_size, void* d_ws, size_t ws_size,
                              hipStream_t stream) {
  (void)in_sizes; (void)n_in; (void)out_size; (void)ws_size;
  const float* v_in  = (const float*)d_in[2];
  const float* qa    = (const float*)d_in[3];
  const float* ka    = (const float*)d_in[4];
  const float* va    = (const float*)d_in[5];
  const float* mask  = (const float*)d_in[8];
  const float* proj  = (const float*)d_in[9];
  const float* enc_w = (const float*)d_in[10];
  const float* enc_b = (const float*)d_in[11];
  const float* ln_g  = (const float*)d_in[12];
  const float* ln_b  = (const float*)d_in[13];
  const float* dec_w = (const float*)d_in[14];
  const float* dec_b = (const float*)d_in[15];
  const float* c1_w  = (const float*)d_in[16];
  const float* c1_b  = (const float*)d_in[17];
  const float* r1a_w = (const float*)d_in[18];
  const float* r1a_b = (const float*)d_in[19];
  const float* r1b_w = (const float*)d_in[20];
  const float* r1b_b = (const float*)d_in[21];
  const float* r2a_w = (const float*)d_in[22];
  const float* r2a_b = (const float*)d_in[23];
  const float* r2b_w = (const float*)d_in[24];
  const float* r2b_b = (const float*)d_in[25];
  const float* c2_w  = (const float*)d_in[26];
  const float* c2_b  = (const float*)d_in[27];

  char* cur = (char*)d_ws;
  auto alloc = [&](size_t bytes) -> void* {
    void* p = (void*)cur;
    cur += (bytes + 255) & ~(size_t)255;
    return p;
  };
  __bf16* qdn   = (__bf16*)alloc((size_t)NHT_ * HID_ * 2);
  __bf16* kdn   = (__bf16*)alloc((size_t)NHT_ * HID_ * 2);
  __bf16* vbf   = (__bf16*)alloc((size_t)NHT_ * HID_ * 2);
  __bf16* vaT   = (__bf16*)alloc((size_t)NH_ * 128 * T_ * 2);
  __bf16* projT = (__bf16*)alloc((size_t)FP_ * HID_ * 2);
  __bf16* encT  = (__bf16*)alloc((size_t)128 * 192 * 2);
  __bf16* decT  = (__bf16*)alloc((size_t)128 * 128 * 2);
  __bf16* wc1   = (__bf16*)alloc((size_t)48 * KP1_ * 2);
  __bf16* w1a   = (__bf16*)alloc((size_t)48 * KP48_ * 2);
  __bf16* w1b   = (__bf16*)alloc((size_t)48 * KP48_ * 2);
  __bf16* w2a   = (__bf16*)alloc((size_t)48 * KP48_ * 2);
  __bf16* w2b   = (__bf16*)alloc((size_t)48 * KP48_ * 2);
  float*  qdiag = (float*)alloc((size_t)NHT_ * 4);
  float*  kdiag = (float*)alloc((size_t)NHT_ * 4);
  float*  iy    = (float*)alloc((size_t)N_ * T_ * 4);
  __bf16* qp    = (__bf16*)alloc((size_t)NHT_ * FP_ * 2);
  __bf16* kpT   = (__bf16*)alloc((size_t)NH_ * FP_ * T_ * 2);
  float*  ksum  = (float*)alloc((size_t)NH_ * FP_ * 4);
  __bf16* ctxT  = (__bf16*)alloc((size_t)NH_ * 128 * FP_ * 2);
  __bf16* pv    = (__bf16*)alloc((size_t)NHT_ * 192 * 2);
  float*  score = (float*)alloc((size_t)NHT_ * 128 * 4);
  __bf16* x1    = (__bf16*)alloc((size_t)N_ * 48 * 1024 * 64 * 2);
  __bf16* ta    = (__bf16*)alloc((size_t)N_ * 48 * 1024 * 64 * 2);
  __bf16* tbuf  = (__bf16*)alloc((size_t)N_ * 48 * 1024 * 64 * 2);

  // prep
  hipLaunchKernelGGL(iy_kernel, dim3(N_), dim3(32), 0, stream, mask, iy);
  hipLaunchKernelGGL(smallprep_kernel, dim3(232), dim3(256), 0, stream,
                     proj, enc_w, dec_w, projT, encT, decT);
  hipLaunchKernelGGL(convw_kernel, dim3(360), dim3(256), 0, stream,
                     c1_w, r1a_w, r1b_w, r2a_w, r2b_w, wc1, w1a, w1b, w2a, w2b);
  hipLaunchKernelGGL(rowprep_kernel, dim3((NHT_ + 255) / 256), dim3(256), 0, stream,
                     qa, ka, v_in, mask, qdn, kdn, vbf, qdiag, kdiag);
  {
    size_t tot = (size_t)NH_ * 128 * T_;
    hipLaunchKernelGGL(vat_kernel, dim3((unsigned)((tot + 255) / 256)), dim3(256), 0, stream,
                       va, mask, iy, vaT);
  }
  // FAVOR+ features, ksum, ctx (async-LDS staged WMMA), out
  hipLaunchKernelGGL(proj_kernel, dim3(NHT_ / 16, 2), dim3(128), 0, stream,
                     qdn, kdn, projT, qdiag, kdiag, qp, kpT);
  hipLaunchKernelGGL(ksum_kernel, dim3(NH_ * FP_), dim3(256), 0, stream, kpT, ksum);
  hipLaunchKernelGGL(ctx_kernel, dim3(NH_, FP_ / 16, 2), dim3(128), 0, stream,
                     kpT, vaT, ctxT);
  hipLaunchKernelGGL(outpv_kernel, dim3(NHT_ / 16), dim3(128), 0, stream,
                     qp, ctxT, ksum, vbf, pv);
  // enc -> LN -> gelu -> dec (WMMA x2)
  hipLaunchKernelGGL(encdec_kernel, dim3(NHT_ / 16), dim3(128), 0, stream,
                     pv, encT, decT, enc_b, ln_g, ln_b, dec_b, score);
  // conv stack: WMMA implicit GEMM, bf16 activations
  {
    dim3 g(N_ * 1024);
    hipLaunchKernelGGL(conv1w_kernel,  g, dim3(128), 0, stream, score, wc1, c1_b, x1);
    hipLaunchKernelGGL(conv48w_kernel, g, dim3(128), 0, stream, x1,  (const __bf16*)nullptr, w1a, r1a_b, ta);
    hipLaunchKernelGGL(conv48w_kernel, g, dim3(128), 0, stream, ta,  x1,                     w1b, r1b_b, tbuf);
    hipLaunchKernelGGL(conv48w_kernel, g, dim3(128), 0, stream, tbuf,(const __bf16*)nullptr, w2a, r2a_b, ta);
    hipLaunchKernelGGL(conv48w_kernel, g, dim3(128), 0, stream, ta,  tbuf,                   w2b, r2b_b, x1);
  }
  // pixel shuffle + c2 + softmax
  hipLaunchKernelGGL(final_kernel, dim3(N_ * 12 * T_), dim3(128), 0, stream,
                     x1, c2_w, c2_b, (float*)d_out);
}
